// MultiheadAttention_695784702349
// MI455X (gfx1250) — matrix operand order repaired and checked by reading, not compiled, not run on hardware
//
#include <hip/hip_runtime.h>

// ---------------------------------------------------------------------------
// MultiheadAttention forward, MI455X (gfx1250), bf16 WMMA + flash softmax.
// Shapes: x[T=2048, B=2, E=1024], H=16 heads, D=64, S=T.
// Workspace layout (48 MB total):
//   [0,   8M)  x_bf16        [4096 x 1024]
//   [8M, 14M)  wqkv_bf16     [3072 x 1024]
//   [14M,16M)  wout_bf16     [1024 x 1024]
//   [16M,24M)  q_scaled      [32 heads][2048][64]   (q * D^-0.5)
//   [24M,32M)  k             [32 heads][2048][64]
//   [32M,40M)  vT            [32 heads][64][2048]   (transposed for P*V)
//   [40M,48M)  ctx_bf16      [4096 x 1024]
// ---------------------------------------------------------------------------

typedef __bf16 bf16;
typedef __attribute__((ext_vector_type(16))) __bf16 v16bf;
typedef __attribute__((ext_vector_type(8)))  __bf16 v8bf;
typedef __attribute__((ext_vector_type(8)))  float  v8f;
typedef __attribute__((ext_vector_type(4)))  int    v4i;

#define T_LEN   2048
#define EMB     1024
#define HD      64
#define NHB     32          // B * H
#define HEADSZ  (2048 * 64) // elements per head buffer
#define SCALING 0.125f      // 64^-0.5

// gfx1250 async global->LDS path (ASYNCcnt), with sync fallback.
#if defined(__has_builtin)
#if __has_builtin(__builtin_amdgcn_global_load_async_to_lds_b128)
#define HAVE_ASYNC_LDS 1
#endif
#endif
#ifndef HAVE_ASYNC_LDS
#define HAVE_ASYNC_LDS 0
#endif

typedef __attribute__((address_space(1))) v4i g_v4i;
typedef __attribute__((address_space(3))) v4i l_v4i;

__device__ __forceinline__ v16bf make_frag(const bf16* lo, const bf16* hi) {
  union { v16bf v; v8bf h[2]; } u;
  u.h[0] = *(const v8bf*)lo;
  u.h[1] = *(const v8bf*)hi;
  return u.v;
}

__device__ __forceinline__ v8f vzero8() {
  return (v8f){0.f, 0.f, 0.f, 0.f, 0.f, 0.f, 0.f, 0.f};
}

__device__ __forceinline__ v8f wmma_bf16(v16bf a, v16bf b, v8f c) {
  return __builtin_amdgcn_wmma_f32_16x16x32_bf16(
      /*neg_a=*/false, a, /*neg_b=*/false, b,
      /*c_mod=*/(short)0, c, /*reuse_a=*/false, /*reuse_b=*/false);
}

// Stage 16 bytes global -> LDS (per lane).
__device__ __forceinline__ void stage16(const bf16* g, bf16* l) {
#if HAVE_ASYNC_LDS
  __builtin_amdgcn_global_load_async_to_lds_b128(
      (g_v4i*)g, (l_v4i*)l, /*offset=*/0, /*cpol=*/0);
#else
  *(v8bf*)l = *(const v8bf*)g;
#endif
}

__device__ __forceinline__ void stage_wait() {
#if HAVE_ASYNC_LDS
#if __has_builtin(__builtin_amdgcn_s_wait_asynccnt)
  __builtin_amdgcn_s_wait_asynccnt(0);
#endif
#endif
}

// ------------------------------ fp32 -> bf16 -------------------------------
__global__ void cvt_bf16_kernel(const float* __restrict__ in,
                                bf16* __restrict__ out, int n) {
  int i = blockIdx.x * blockDim.x + threadIdx.x;
  int stride = gridDim.x * blockDim.x;
  for (; i < n; i += stride) out[i] = (bf16)in[i];
}

// ------------------------- QKV projection GEMM -----------------------------
// One wave (32 threads) computes a 32x64 tile of qkv = x * Wqkv^T + bias.
// Epilogue scatters into per-head q_scaled / k / vT buffers.
__global__ void qkv_gemm_kernel(const bf16* __restrict__ xb,
                                const bf16* __restrict__ wb,
                                const float* __restrict__ bias,
                                bf16* __restrict__ qs,
                                bf16* __restrict__ kb,
                                bf16* __restrict__ vt) {
  const int lane = threadIdx.x;
  const int ml   = lane & 15;
  const int half = lane >> 4;
  const int m0   = blockIdx.x * 32;   // rows of x (t*B+b flattened)
  const int n0   = blockIdx.y * 64;   // rows of Wqkv (output feature)

  v8f acc[2][4];
#pragma unroll
  for (int i = 0; i < 2; ++i)
#pragma unroll
    for (int j = 0; j < 4; ++j) acc[i][j] = vzero8();

  for (int k0 = 0; k0 < EMB; k0 += 32) {
    v16bf a[2], b[4];
#pragma unroll
    for (int i = 0; i < 2; ++i) {
      const bf16* p = xb + (size_t)(m0 + i * 16 + ml) * EMB + k0 + half * 8;
      a[i] = make_frag(p, p + 16);
    }
#pragma unroll
    for (int j = 0; j < 4; ++j) {
      const bf16* p = wb + (size_t)(n0 + j * 16 + ml) * EMB + k0 + half * 8;
      b[j] = make_frag(p, p + 16);
    }
#pragma unroll
    for (int i = 0; i < 2; ++i)
#pragma unroll
      for (int j = 0; j < 4; ++j) acc[i][j] = wmma_bf16(a[i], b[j], acc[i][j]);
  }

  const int sec = n0 >> 10;        // 0=q, 1=k, 2=v
  const int c0  = n0 & 1023;       // column within section (mult of 64)
  const int h   = c0 >> 6;         // head
  float bv[4];
#pragma unroll
  for (int j = 0; j < 4; ++j) bv[j] = bias[n0 + j * 16 + ml];

#pragma unroll
  for (int i = 0; i < 2; ++i)
#pragma unroll
    for (int j = 0; j < 4; ++j)
#pragma unroll
      for (int r = 0; r < 8; ++r) {
        int gm = m0 + i * 16 + r + 8 * half;     // flattened t*B+b
        int t  = gm >> 1, bz = gm & 1;
        int d  = j * 16 + ml;
        int hb = bz * 16 + h;                    // b*H + h
        float v = acc[i][j][r] + bv[j];
        if (sec == 0) {
          qs[(size_t)hb * HEADSZ + (size_t)t * HD + d] = (bf16)(v * SCALING);
        } else if (sec == 1) {
          kb[(size_t)hb * HEADSZ + (size_t)t * HD + d] = (bf16)v;
        } else {
          vt[(size_t)hb * HEADSZ + (size_t)d * T_LEN + t] = (bf16)v;
        }
      }
}

// ------------------------- fused flash attention ---------------------------
// 4 waves per workgroup; each wave owns 16 query rows (block = 64 rows).
// Per 64-key step: K tile (64x64) and vT tile (64x64) are staged once into
// LDS cooperatively (async global->LDS when available) and shared by all 4
// waves, cutting VMEM traffic 4x. Online softmax; P relaid C->A layout via
// per-wave padded LDS tile.
#define P_STRIDE  72
#define KV_STRIDE 72
__global__ __launch_bounds__(128) void attn_kernel(
    const bf16* __restrict__ qs_all,
    const bf16* __restrict__ kb_all,
    const bf16* __restrict__ vt_all,
    bf16* __restrict__ ctx) {
  __shared__ __align__(16) bf16 ktile[64 * KV_STRIDE];
  __shared__ __align__(16) bf16 vtile[64 * KV_STRIDE];
  __shared__ __align__(16) bf16 ptile[4][16 * P_STRIDE];

  const int tid  = threadIdx.x;
  const int lane = tid & 31;
  const int wave = tid >> 5;
  const int ml   = lane & 15;
  const int half = (lane >> 4) & 1;
  const int hb   = blockIdx.y;
  const int q0   = blockIdx.x * 64 + wave * 16;

  const bf16* qs = qs_all + (size_t)hb * HEADSZ;
  const bf16* kb = kb_all + (size_t)hb * HEADSZ;
  const bf16* vt = vt_all + (size_t)hb * HEADSZ;

  v16bf aq[2];
#pragma unroll
  for (int kk = 0; kk < 2; ++kk) {
    const bf16* p = qs + (size_t)(q0 + ml) * HD + kk * 32 + half * 8;
    aq[kk] = make_frag(p, p + 16);
  }

  v8f o[4];
#pragma unroll
  for (int dj = 0; dj < 4; ++dj) o[dj] = vzero8();
  float mrow[8], lrow[8];
#pragma unroll
  for (int r = 0; r < 8; ++r) { mrow[r] = -3.0e38f; lrow[r] = 0.f; }

  for (int s0 = 0; s0 < T_LEN; s0 += 64) {
    // ---- cooperative staging of K and vT tiles (64x64 bf16 each) ----
#pragma unroll
    for (int c = 0; c < 4; ++c) {
      int id  = c * 128 + tid;     // 0..511 16-byte chunks per tile
      int row = id >> 3;           // 0..63
      int col = (id & 7) * 8;      // element offset within row
      stage16(kb + (size_t)(s0 + row) * HD + col, &ktile[row * KV_STRIDE + col]);
      stage16(vt + (size_t)row * T_LEN + s0 + col, &vtile[row * KV_STRIDE + col]);
    }
    stage_wait();
    __syncthreads();

    // ---- scores tile S = Q * K^T  (16 x 64, fp32) from LDS ----
    v8f sc[4];
#pragma unroll
    for (int j = 0; j < 4; ++j) {
      sc[j] = vzero8();
#pragma unroll
      for (int kk = 0; kk < 2; ++kk) {
        const bf16* p = &ktile[(j * 16 + ml) * KV_STRIDE + kk * 32 + half * 8];
        v16bf bk = make_frag(p, p + 16);
        sc[j] = wmma_bf16(aq[kk], bk, sc[j]);
      }
    }

    // ---- online softmax (rows live across 16-lane halves) ----
#pragma unroll
    for (int r = 0; r < 8; ++r) {
      float mx = fmaxf(fmaxf(sc[0][r], sc[1][r]), fmaxf(sc[2][r], sc[3][r]));
      mx = fmaxf(mx, __shfl_xor(mx, 1));
      mx = fmaxf(mx, __shfl_xor(mx, 2));
      mx = fmaxf(mx, __shfl_xor(mx, 4));
      mx = fmaxf(mx, __shfl_xor(mx, 8));
      float mnew  = fmaxf(mrow[r], mx);
      float scale = __expf(mrow[r] - mnew);
      float rs = 0.f;
#pragma unroll
      for (int j = 0; j < 4; ++j) {
        float p = __expf(sc[j][r] - mnew);
        rs += p;
        ptile[wave][(r + 8 * half) * P_STRIDE + j * 16 + ml] = (bf16)p;
      }
      rs += __shfl_xor(rs, 1);
      rs += __shfl_xor(rs, 2);
      rs += __shfl_xor(rs, 4);
      rs += __shfl_xor(rs, 8);
      lrow[r] = lrow[r] * scale + rs;
      mrow[r] = mnew;
#pragma unroll
      for (int dj = 0; dj < 4; ++dj) o[dj][r] *= scale;
    }

    // ---- relayout P (C-layout, per-wave LDS) into A-fragments ----
    v16bf ap[2];
#pragma unroll
    for (int kk = 0; kk < 2; ++kk) {
      const bf16* p = &ptile[wave][ml * P_STRIDE + kk * 32 + half * 8];
      ap[kk] = make_frag(p, p + 16);
    }

    // ---- O += P * V  (vT tile in LDS as row-major-K "weight") ----
#pragma unroll
    for (int dj = 0; dj < 4; ++dj)
#pragma unroll
      for (int kk = 0; kk < 2; ++kk) {
        const bf16* p = &vtile[(dj * 16 + ml) * KV_STRIDE + kk * 32 + half * 8];
        v16bf bv = make_frag(p, p + 16);
        o[dj] = wmma_bf16(ap[kk], bv, o[dj]);
      }
    __syncthreads();   // protect ktile/vtile before next stage
  }

  // ---- normalize and scatter into ctx [t*B+b][h*64+d] as bf16 ----
  const int h  = hb & 15;
  const int bz = hb >> 4;
#pragma unroll
  for (int dj = 0; dj < 4; ++dj)
#pragma unroll
    for (int r = 0; r < 8; ++r) {
      int t   = q0 + r + 8 * half;
      int row = t * 2 + bz;
      int col = h * HD + dj * 16 + ml;
      ctx[(size_t)row * EMB + col] = (bf16)(o[dj][r] / lrow[r]);
    }
}

// --------------------------- output projection -----------------------------
__global__ void out_gemm_kernel(const bf16* __restrict__ ctx,
                                const bf16* __restrict__ wb,
                                const float* __restrict__ bias,
                                float* __restrict__ out) {
  const int lane = threadIdx.x;
  const int ml   = lane & 15;
  const int half = lane >> 4;
  const int m0   = blockIdx.x * 32;
  const int n0   = blockIdx.y * 64;

  v8f acc[2][4];
#pragma unroll
  for (int i = 0; i < 2; ++i)
#pragma unroll
    for (int j = 0; j < 4; ++j) acc[i][j] = vzero8();

  for (int k0 = 0; k0 < EMB; k0 += 32) {
    v16bf a[2], b[4];
#pragma unroll
    for (int i = 0; i < 2; ++i) {
      const bf16* p = ctx + (size_t)(m0 + i * 16 + ml) * EMB + k0 + half * 8;
      a[i] = make_frag(p, p + 16);
    }
#pragma unroll
    for (int j = 0; j < 4; ++j) {
      const bf16* p = wb + (size_t)(n0 + j * 16 + ml) * EMB + k0 + half * 8;
      b[j] = make_frag(p, p + 16);
    }
#pragma unroll
    for (int i = 0; i < 2; ++i)
#pragma unroll
      for (int j = 0; j < 4; ++j) acc[i][j] = wmma_bf16(a[i], b[j], acc[i][j]);
  }

  float bv[4];
#pragma unroll
  for (int j = 0; j < 4; ++j) bv[j] = bias[n0 + j * 16 + ml];

#pragma unroll
  for (int i = 0; i < 2; ++i)
#pragma unroll
    for (int j = 0; j < 4; ++j)
#pragma unroll
      for (int r = 0; r < 8; ++r) {
        int gm = m0 + i * 16 + r + 8 * half;
        int gn = n0 + j * 16 + ml;
        out[(size_t)gm * EMB + gn] = acc[i][j][r] + bv[j];
      }
}

// ---------------------------------------------------------------------------
extern "C" void kernel_launch(void* const* d_in, const int* in_sizes, int n_in,
                              void* d_out, int out_size, void* d_ws, size_t ws_size,
                              hipStream_t stream) {
  const float* x      = (const float*)d_in[0];   // [2048, 2, 1024]
  const float* w_in   = (const float*)d_in[1];   // [3072, 1024]
  const float* b_in   = (const float*)d_in[2];   // [3072]
  const float* w_out  = (const float*)d_in[3];   // [1024, 1024]
  const float* b_out  = (const float*)d_in[4];   // [1024]
  float*       out    = (float*)d_out;           // [2048, 2, 1024]

  char* ws = (char*)d_ws;
  const size_t MB = 1024 * 1024;
  bf16* xb   = (bf16*)(ws + 0 * MB);
  bf16* wqkv = (bf16*)(ws + 8 * MB);
  bf16* wout = (bf16*)(ws + 14 * MB);
  bf16* qs   = (bf16*)(ws + 16 * MB);
  bf16* kb   = (bf16*)(ws + 24 * MB);
  bf16* vt   = (bf16*)(ws + 32 * MB);
  bf16* ctx  = (bf16*)(ws + 40 * MB);
  if (ws_size < 48 * MB) return;  // workspace contract: need 48 MB

  // 1) fp32 -> bf16 conversions
  cvt_bf16_kernel<<<4096, 256, 0, stream>>>(x, xb, 4096 * 1024);
  cvt_bf16_kernel<<<4096, 256, 0, stream>>>(w_in, wqkv, 3072 * 1024);
  cvt_bf16_kernel<<<2048, 256, 0, stream>>>(w_out, wout, 1024 * 1024);

  // 2) fused QKV projection -> per-head q(scaled), k, vT
  qkv_gemm_kernel<<<dim3(4096 / 32, 3072 / 64), 32, 0, stream>>>(
      xb, wqkv, b_in, qs, kb, vt);

  // 3) fused flash attention -> ctx bf16 [4096, 1024]
  attn_kernel<<<dim3(T_LEN / 64, NHB), 128, 0, stream>>>(qs, kb, vt, ctx);

  // 4) output projection (fp32 out)
  out_gemm_kernel<<<dim3(4096 / 32, 1024 / 64), 32, 0, stream>>>(
      ctx, wout, b_out, out);
}
